// LiquidTransformer_16303695855793
// MI455X (gfx1250) — matrix operand order repaired and checked by reading, not compiled, not run on hardware
//
#include <hip/hip_runtime.h>
#include <math.h>

// ---------------------------------------------------------------------------
// LiquidTransformer forward for MI455X (gfx1250, wave32, WMMA).
// All large GEMMs use v_wmma_f32_16x16x32_f16 (f16 in / f32 accumulate).
// Fragment operands are laid out so every load is a contiguous b128:
//   A fragments: row-major, two v8h loads per lane.
//   B fragments: column-major-within-K-tile, two v8h loads per lane.
// GEMM k-loop is double-buffered (stage k+1 while computing k, one barrier
// per iteration); staging gathers all global b128 loads into registers
// before converting (one loadcnt wait per tile, not per load) and
// prefetches two K-tiles ahead (global_prefetch_b8).
// All GEMM shapes are exact tile multiples (M%64==0, N%128==0, K%32==0) and
// all staging loops have fixed trip counts => no exec-mask juggling.
// Weights total ~140MB f32 -> L2 (192MB) resident; GEMMs are WMMA-bound.
// Workspace assumes ws_size >= ~64 MB.
// ---------------------------------------------------------------------------

typedef __attribute__((ext_vector_type(16))) _Float16 v16h;
typedef __attribute__((ext_vector_type(8)))  _Float16 v8h;
typedef __attribute__((ext_vector_type(4)))  _Float16 v4h;
typedef __attribute__((ext_vector_type(8)))  float    v8f;

constexpr int Bb  = 8;
constexpr int Ss  = 256;
constexpr int Hh  = 512;
constexpr int Vv  = 32000;
constexpr int NHh = 8;
constexpr int Ee  = 4;
constexpr int Ff  = 2048;
constexpr int BUu = 64;
constexpr float LNEPS = 1e-5f;

__device__ inline v8f wmma_f16(v16h a, v16h b, v8f c) {
  // D = A(16x32 f16) * B(32x16 f16) + C(16x16 f32)
  return __builtin_amdgcn_wmma_f32_16x16x32_f16(
      false, a, false, b, (short)0, c, false, false);
}

__device__ inline v16h cat8(v8h lo, v8h hi) {
  return __builtin_shufflevector(lo, hi, 0, 1, 2, 3, 4, 5, 6, 7,
                                 8, 9, 10, 11, 12, 13, 14, 15);
}

__device__ inline v4h cvt4(float4 f) {
  v4h h;
  h[0] = (_Float16)f.x; h[1] = (_Float16)f.y;
  h[2] = (_Float16)f.z; h[3] = (_Float16)f.w;
  return h;
}

// A fragment (16x32 f16) from row-major [M x K], ld multiple of 8.
// Lanes 0-15: M=lane, K pairs {hf*8..+7, 16+hf*8..+7} with hf=0;
// lanes 16-31: same rows, hf=1 (per CDNA5 ISA 7.12.2).  Two b128 loads.
__device__ inline v16h fragA(const _Float16* A, int lda, int m0, int k0) {
  const int lane = threadIdx.x & 31;
  const int m  = m0 + (lane & 15);
  const int hf = lane >> 4;
  const _Float16* p = A + m * lda + k0 + hf * 8;
  return cat8(*(const v8h*)p, *(const v8h*)(p + 16));
}

// Same, but only 8 valid rows exist (rows 8-15 replicate 0-7; outputs for
// those rows are discarded).  Used by the B=8 LTC scan.
__device__ inline v16h fragA8(const _Float16* A, int lda, int k0) {
  const int lane = threadIdx.x & 31;
  const int m  = lane & 7;
  const int hf = (lane >> 4) & 1;
  const _Float16* p = A + m * lda + k0 + hf * 8;
  return cat8(*(const v8h*)p, *(const v8h*)(p + 16));
}

// B fragment (32x16 f16) from column-major storage Bcm[n * ldk + k].
// Lane reads K {hf*8..+7, 16+hf*8..+7} of its column: two b128 loads.
__device__ inline v16h fragB(const _Float16* Bcm, int ldk, int k0, int n0) {
  const int lane = threadIdx.x & 31;
  const int n  = n0 + (lane & 15);
  const int hf = lane >> 4;
  const _Float16* p = Bcm + n * ldk + k0 + hf * 8;
  return cat8(*(const v8h*)p, *(const v8h*)(p + 16));
}

// Stage one 64x32 A tile (row-major f16) and one 32x128 B tile
// (column-major f16) into LDS, converting f32 -> f16.  All loads are
// gathered into registers first so several b128 loads stay in flight.
// Fixed trip counts, no guards.  Prefetch two K-tiles ahead.
__device__ inline void stage_tiles(const float* __restrict__ A,
                                   const float* __restrict__ Bm,
                                   _Float16* __restrict__ dA,
                                   _Float16* __restrict__ dB,
                                   int N, int K, int mBase, int nBase, int k0,
                                   int b_trans, int tid) {
  float4 fa[2];
#pragma unroll
  for (int it = 0; it < 2; ++it) {
    const int i = tid * 4 + it * 1024;
    const int r = i >> 5, c = i & 31;
    const float* p = A + (mBase + r) * K + k0 + c;
    fa[it] = *(const float4*)p;
    __builtin_prefetch(p + 64, 0, 3);  // two K-tiles ahead, near-cache
  }
  float4 fb[4];
  if (!b_trans) {
    // B[K,N]: coalesced-on-n float4 reads.
#pragma unroll
    for (int it = 0; it < 4; ++it) {
      const int i = tid * 4 + it * 1024;
      const int r = i >> 7, c = i & 127;
      const float* p = Bm + (k0 + r) * N + nBase + c;
      fb[it] = *(const float4*)p;
      __builtin_prefetch(p + 64 * N, 0, 3);
    }
  } else {
    // B[N,K]: contiguous along K.
#pragma unroll
    for (int it = 0; it < 4; ++it) {
      const int i = tid * 4 + it * 1024;
      const int kl = i & 31, c = i >> 5;
      const float* p = Bm + (nBase + c) * K + k0 + kl;
      fb[it] = *(const float4*)p;
      __builtin_prefetch(p + 64, 0, 3);
    }
  }
  // convert + store after all loads are issued
#pragma unroll
  for (int it = 0; it < 2; ++it) {
    const int i = tid * 4 + it * 1024;
    *(v4h*)(dA + i) = cvt4(fa[it]);
  }
  if (!b_trans) {
#pragma unroll
    for (int it = 0; it < 4; ++it) {
      const int i = tid * 4 + it * 1024;
      const int r = i >> 7, c = i & 127;
      dB[(c + 0) * 32 + r] = (_Float16)fb[it].x;
      dB[(c + 1) * 32 + r] = (_Float16)fb[it].y;
      dB[(c + 2) * 32 + r] = (_Float16)fb[it].z;
      dB[(c + 3) * 32 + r] = (_Float16)fb[it].w;
    }
  } else {
#pragma unroll
    for (int it = 0; it < 4; ++it) {
      const int i = tid * 4 + it * 1024;
      const int kl = i & 31, c = i >> 5;
      *(v4h*)(dB + c * 32 + kl) = cvt4(fb[it]);
    }
  }
}

// ---------------------------------------------------------------------------
// Generic WMMA GEMM: C[M,N] = epilogue(A[M,K] @ B[K,N] + bias)
// Block tile 64x128, BK=32, double-buffered, 8 waves x (32x32) sub-tiles.
// act: 0=none 1=sigmoid 2=gelu(exact) 3=tanh
// row_scale: optional per-row multiplier (stride rs_stride), accum: C += val
// b_trans: B is stored [N,K] row-major (tied lm_head).
// ---------------------------------------------------------------------------
__global__ __launch_bounds__(256) void wmma_gemm_kernel(
    const float* __restrict__ A, const float* __restrict__ Bm,
    const float* __restrict__ bias, float* __restrict__ C,
    int M, int N, int K, int act,
    const float* __restrict__ row_scale, int rs_stride, int accum, int b_trans) {
  __shared__ __align__(32) _Float16 sA[2][64 * 32];   // row-major [64][32]
  __shared__ __align__(32) _Float16 sB[2][128 * 32];  // column-major [n][k]

  const int tid   = threadIdx.x;
  const int wid   = tid >> 5;
  const int lane  = tid & 31;
  const int mBase = blockIdx.y * 64;
  const int nBase = blockIdx.x * 128;
  const int wm    = (wid & 1) * 32;   // 2 wave rows
  const int wn    = (wid >> 1) * 32;  // 4 wave cols

  v8f acc[2][2];
#pragma unroll
  for (int i = 0; i < 2; ++i)
#pragma unroll
    for (int j = 0; j < 2; ++j)
#pragma unroll
      for (int v = 0; v < 8; ++v) acc[i][j][v] = 0.f;

  stage_tiles(A, Bm, sA[0], sB[0], N, K, mBase, nBase, 0, b_trans, tid);
  __syncthreads();

  int cur = 0;
  for (int k0 = 0; k0 < K; k0 += 32) {
    const int nxt = cur ^ 1;
    if (k0 + 32 < K)  // stage next tile while computing current
      stage_tiles(A, Bm, sA[nxt], sB[nxt], N, K, mBase, nBase, k0 + 32,
                  b_trans, tid);

    const v16h a0 = fragA(sA[cur], 32, wm, 0);
    const v16h a1 = fragA(sA[cur], 32, wm + 16, 0);
    const v16h b0 = fragB(sB[cur], 32, 0, wn);
    const v16h b1 = fragB(sB[cur], 32, 0, wn + 16);
    acc[0][0] = wmma_f16(a0, b0, acc[0][0]);
    acc[0][1] = wmma_f16(a0, b1, acc[0][1]);
    acc[1][0] = wmma_f16(a1, b0, acc[1][0]);
    acc[1][1] = wmma_f16(a1, b1, acc[1][1]);
    __syncthreads();
    cur = nxt;
  }

  const int nl = lane & 15;
  const int mb = (lane >> 4) * 8;  // C/D: lanes 16-31 hold rows M=8..15
#pragma unroll
  for (int i = 0; i < 2; ++i)
#pragma unroll
    for (int j = 0; j < 2; ++j) {
      const int n = nBase + wn + j * 16 + nl;
      const float bv = bias ? bias[n] : 0.f;
#pragma unroll
      for (int v = 0; v < 8; ++v) {
        const int m = mBase + wm + i * 16 + mb + v;
        float val = acc[i][j][v] + bv;
        if (act == 1)      val = 1.f / (1.f + expf(-val));
        else if (act == 2) val = 0.5f * val * (1.f + erff(val * 0.70710678f));
        else if (act == 3) val = tanhf(val);
        if (row_scale) val *= row_scale[m * rs_stride];
        const int o = m * N + n;
        if (accum) C[o] += val; else C[o] = val;
      }
    }
}

// ---------------------------------------------------------------------------
// Embedding gather: x[b,s,:] = tok_emb[id] + pos_emb[s]
// Launched with exactly T*H threads.
// ---------------------------------------------------------------------------
__global__ void embed_kernel(const int* __restrict__ ids,
                             const float* __restrict__ tok,
                             const float* __restrict__ pos,
                             float* __restrict__ x) {
  const int idx = blockIdx.x * 256 + threadIdx.x;
  const int c = idx % Hh;
  const int t = idx / Hh;
  const int s = t % Ss;
  x[idx] = tok[ids[t] * Hh + c] + pos[s * Hh + c];
}

__global__ void zero_kernel(float* __restrict__ p, int n) {
  const int i = blockIdx.x * 256 + threadIdx.x;
  if (i < n) p[i] = 0.f;
}

// Convert [K,N] f32 weight -> [N,K] f16 (column-major) for b128 frag loads.
// Launched with exactly K*N threads.
__global__ void conv_w_kernel(const float* __restrict__ W,
                              _Float16* __restrict__ out, int K, int N) {
  const int idx = blockIdx.x * 256 + threadIdx.x;
  const int k = idx / N, n = idx % N;
  out[n * K + k] = (_Float16)W[idx];
}

// ---------------------------------------------------------------------------
// LayerNorm over H=512 with optional residual; one block per token.
// ---------------------------------------------------------------------------
__global__ __launch_bounds__(256) void ln_kernel(
    const float* __restrict__ in, const float* __restrict__ res,
    const float* __restrict__ g, const float* __restrict__ b,
    float* __restrict__ out) {
  __shared__ float red[256];
  const int t = blockIdx.x, tid = threadIdx.x;
  const int base = t * Hh;
  float v0 = in[base + tid]       + (res ? res[base + tid]       : 0.f);
  float v1 = in[base + 256 + tid] + (res ? res[base + 256 + tid] : 0.f);
  red[tid] = v0 + v1;
  __syncthreads();
  for (int s = 128; s > 0; s >>= 1) {
    if (tid < s) red[tid] += red[tid + s];
    __syncthreads();
  }
  const float mu = red[0] / (float)Hh;
  __syncthreads();
  const float d0 = v0 - mu, d1 = v1 - mu;
  red[tid] = d0 * d0 + d1 * d1;
  __syncthreads();
  for (int s = 128; s > 0; s >>= 1) {
    if (tid < s) red[tid] += red[tid + s];
    __syncthreads();
  }
  const float rstd = rsqrtf(red[0] / (float)Hh + LNEPS);
  out[base + tid]       = d0 * rstd * g[tid]       + b[tid];
  out[base + 256 + tid] = d1 * rstd * g[256 + tid] + b[256 + tid];
}

// ---------------------------------------------------------------------------
// MoE gate: softmax over 4 experts, top-2, normalized combine weights.
// Launched with exactly T threads.
// ---------------------------------------------------------------------------
__global__ void moe_gate_kernel(const float* __restrict__ x,
                                const float* __restrict__ gw,
                                const float* __restrict__ gb,
                                float* __restrict__ wgt) {
  const int m = blockIdx.x * 256 + threadIdx.x;
  float lg[Ee];
#pragma unroll
  for (int e = 0; e < Ee; ++e) lg[e] = gb[e];
  for (int kk = 0; kk < Hh; ++kk) {
    const float xv = x[m * Hh + kk];
#pragma unroll
    for (int e = 0; e < Ee; ++e) lg[e] += xv * gw[kk * Ee + e];
  }
  float mx = lg[0];
#pragma unroll
  for (int e = 1; e < Ee; ++e) mx = fmaxf(mx, lg[e]);
  float p[Ee], sum = 0.f;
#pragma unroll
  for (int e = 0; e < Ee; ++e) { p[e] = expf(lg[e] - mx); sum += p[e]; }
#pragma unroll
  for (int e = 0; e < Ee; ++e) p[e] /= sum;
  int i1 = 0;
#pragma unroll
  for (int e = 1; e < Ee; ++e) if (p[e] > p[i1]) i1 = e;
  int i2 = -1;
#pragma unroll
  for (int e = 0; e < Ee; ++e)
    if (e != i1 && (i2 < 0 || p[e] > p[i2])) i2 = e;
  const float tot = p[i1] + p[i2];
#pragma unroll
  for (int e = 0; e < Ee; ++e) wgt[m * Ee + e] = 0.f;
  wgt[m * Ee + i1] = p[i1] / tot;
  wgt[m * Ee + i2] = p[i2] / tot;
}

// ---------------------------------------------------------------------------
// Causal attention, flash-style.  Block = (64-query tile, head, batch).
// Q*K^T and P*V via WMMA; online softmax; O fragments live in registers.
// sK is [key][d] (column-major along the d-contraction => natural layout);
// sVt is [d][key] (column-major along the key-contraction).
// ---------------------------------------------------------------------------
__global__ __launch_bounds__(256) void attn_kernel(
    const float* __restrict__ Q, const float* __restrict__ Kt,
    const float* __restrict__ Vt, float* __restrict__ O) {
  __shared__ __align__(32) _Float16 sQ[64 * 64];   // [q][d], pre-scaled
  __shared__ __align__(32) _Float16 sK[64 * 64];   // [key][d]
  __shared__ __align__(32) _Float16 sVt[64 * 64];  // [d][key]
  __shared__ __align__(32) _Float16 sP[64 * 64];   // probs f16 [q][key]
  __shared__ float sS[64 * 64];                    // scores f32
  __shared__ float sM[64], sL[64], sScale[64];

  const int tid = threadIdx.x, lane = tid & 31, wid = tid >> 5;
  const int qt = blockIdx.x, h = blockIdx.y, b = blockIdx.z;
  const int q0 = qt * 64;
  const int wm = (wid & 3) * 16;    // 4 row groups of 16 queries
  const int wn = (wid >> 2) * 32;   // 2 col groups of 32

  {
    float4 f[4];
#pragma unroll
    for (int it = 0; it < 4; ++it) {
      const int i = tid * 4 + it * 1024;
      const int r = i >> 6, d = i & 63;
      f[it] = *(const float4*)(Q + (b * Ss + q0 + r) * Hh + h * 64 + d);
    }
#pragma unroll
    for (int it = 0; it < 4; ++it) {
      const int i = tid * 4 + it * 1024;
      v4h hh;
      hh[0] = (_Float16)(0.125f * f[it].x); hh[1] = (_Float16)(0.125f * f[it].y);
      hh[2] = (_Float16)(0.125f * f[it].z); hh[3] = (_Float16)(0.125f * f[it].w);
      *(v4h*)(sQ + i) = hh;
    }
  }
  if (tid < 64) { sM[tid] = -1e30f; sL[tid] = 0.f; }

  v8f o0, o1;
#pragma unroll
  for (int v = 0; v < 8; ++v) { o0[v] = 0.f; o1[v] = 0.f; }

  for (int kc = 0; kc <= q0; kc += 64) {
    {
      float4 fk[4], fv[4];
#pragma unroll
      for (int it = 0; it < 4; ++it) {
        const int i = tid * 4 + it * 1024;
        const int ky = i >> 6, d = i & 63;
        fk[it] = *(const float4*)(Kt + (b * Ss + kc + ky) * Hh + h * 64 + d);
        fv[it] = *(const float4*)(Vt + (b * Ss + kc + ky) * Hh + h * 64 + d);
      }
#pragma unroll
      for (int it = 0; it < 4; ++it) {
        const int i = tid * 4 + it * 1024;
        *(v4h*)(sK + i) = cvt4(fk[it]);
      }
#pragma unroll
      for (int it = 0; it < 4; ++it) {
        const int i = tid * 4 + it * 1024;
        const int ky = i >> 6, d = i & 63;  // store transposed
        sVt[(d + 0) * 64 + ky] = (_Float16)fv[it].x;
        sVt[(d + 1) * 64 + ky] = (_Float16)fv[it].y;
        sVt[(d + 2) * 64 + ky] = (_Float16)fv[it].z;
        sVt[(d + 3) * 64 + ky] = (_Float16)fv[it].w;
      }
    }
    __syncthreads();

    // scores chunk S[64q][64k] = Q @ K^T (contraction over d)
    v8f s0, s1;
#pragma unroll
    for (int v = 0; v < 8; ++v) { s0[v] = 0.f; s1[v] = 0.f; }
#pragma unroll
    for (int ks = 0; ks < 2; ++ks) {
      const v16h a = fragA(sQ, 64, wm, ks * 32);
      s0 = wmma_f16(a, fragB(sK, 64, ks * 32, wn), s0);
      s1 = wmma_f16(a, fragB(sK, 64, ks * 32, wn + 16), s1);
    }
    {
      const int nl = lane & 15, mb = (lane >> 4) * 8;
#pragma unroll
      for (int v = 0; v < 8; ++v) {
        sS[(wm + mb + v) * 64 + wn + nl]      = s0[v];
        sS[(wm + mb + v) * 64 + wn + 16 + nl] = s1[v];
      }
    }
    __syncthreads();

    // online softmax update, one thread per query row
    if (tid < 64) {
      const int r = tid, qg = q0 + r;
      float mx = sM[r];
      for (int c = 0; c < 64; ++c) {
        const float sc = ((kc + c) <= qg) ? sS[r * 64 + c] : -1e30f;
        mx = fmaxf(mx, sc);
      }
      const float scale = expf(sM[r] - mx);
      float sum = 0.f;
      for (int c = 0; c < 64; ++c) {
        const float p = ((kc + c) <= qg) ? expf(sS[r * 64 + c] - mx) : 0.f;
        sP[r * 64 + c] = (_Float16)p;
        sum += p;
      }
      sL[r] = sL[r] * scale + sum;
      sM[r] = mx;
      sScale[r] = scale;
    }
    __syncthreads();

    // rescale running O and accumulate P @ V (contraction over key)
    {
      const int mb = (lane >> 4) * 8;
#pragma unroll
      for (int v = 0; v < 8; ++v) {
        const float sc = sScale[wm + mb + v];
        o0[v] *= sc;
        o1[v] *= sc;
      }
    }
#pragma unroll
    for (int ks = 0; ks < 2; ++ks) {
      const v16h a = fragA(sP, 64, wm, ks * 32);
      o0 = wmma_f16(a, fragB(sVt, 64, ks * 32, wn), o0);
      o1 = wmma_f16(a, fragB(sVt, 64, ks * 32, wn + 16), o1);
    }
    __syncthreads();
  }

  {
    const int nl = lane & 15, mb = (lane >> 4) * 8;
#pragma unroll
    for (int v = 0; v < 8; ++v) {
      const int r = wm + mb + v;
      const float inv = 1.f / sL[r];
      const int base = (b * Ss + q0 + r) * Hh + h * 64;
      O[base + wn + nl]      = o0[v] * inv;
      O[base + wn + 16 + nl] = o1[v] * inv;
    }
  }
}

// ---------------------------------------------------------------------------
// LTC sequential scan (one persistent workgroup, 8 waves).
// Per step t, 6 ODE unfolds.  Stage 1: waves 0-3 tau branch, waves 4-7
// backbone ([16,1024]x[1024,64], 32 WMMA k-steps, weights pre-converted to
// f16 column-major => 2 global b128 per fragment, L2-resident across 1536
// reuses).  Stage 2: [16,64]x[64,512]; backbone waves keep fragments in
// registers and apply the ODE update against LDS-resident state.
// Rows 8-15 of every fragment are replicated garbage and never stored.
// ---------------------------------------------------------------------------
__global__ __launch_bounds__(256) void ltc_scan_kernel(
    const float* __restrict__ x, const float* __restrict__ g,
    const _Float16* __restrict__ tau1_wc, const float* __restrict__ tau1_b,
    const _Float16* __restrict__ tau2_wc, const float* __restrict__ tau2_b,
    const _Float16* __restrict__ bb1_wc,  const float* __restrict__ bb1_b,
    const _Float16* __restrict__ bb2_wc,  const float* __restrict__ bb2_b,
    float* __restrict__ ys) {
  __shared__ __align__(32) _Float16 sComb[8 * 1024];  // [b][x_t | s] f16
  __shared__ __align__(32) _Float16 sT1[8 * 64];
  __shared__ __align__(32) _Float16 sF1[8 * 64];
  __shared__ float sTau[8 * 512];
  __shared__ float sState[8 * 512];

  const int tid = threadIdx.x, lane = tid & 31, wid = tid >> 5;
  const float dt = 1.f / 6.f;

#pragma unroll
  for (int it = 0; it < 16; ++it) sState[tid + it * 256] = 0.f;
  __syncthreads();

  for (int t = 0; t < Ss; ++t) {
    {
      float4 f[4];
#pragma unroll
      for (int it = 0; it < 4; ++it) {
        const int i = tid * 4 + it * 1024;
        const int bi = i >> 9, c = i & 511;
        f[it] = *(const float4*)(x + (bi * Ss + t) * Hh + c);
      }
#pragma unroll
      for (int it = 0; it < 4; ++it) {
        const int i = tid * 4 + it * 1024;
        const int bi = i >> 9, c = i & 511;
        *(v4h*)(sComb + bi * 1024 + c) = cvt4(f[it]);
      }
    }
    for (int u = 0; u < 6; ++u) {
#pragma unroll
      for (int it = 0; it < 4; ++it) {
        const int i = tid * 4 + it * 1024;
        const int bi = i >> 9, c = i & 511;
        v4h hh;
        hh[0] = (_Float16)sState[i + 0];
        hh[1] = (_Float16)sState[i + 1];
        hh[2] = (_Float16)sState[i + 2];
        hh[3] = (_Float16)sState[i + 3];
        *(v4h*)(sComb + bi * 1024 + 512 + c) = hh;
      }
      __syncthreads();

      // stage 1: t1 = tanh(comb @ W1 + b1), N=64 -> 4 fragments per branch
      {
        const _Float16* W = (wid < 4) ? tau1_wc : bb1_wc;
        const float* Bv   = (wid < 4) ? tau1_b  : bb1_b;
        const int n0 = (wid & 3) * 16;
        v8f acc;
#pragma unroll
        for (int v = 0; v < 8; ++v) acc[v] = 0.f;
        for (int k0 = 0; k0 < 1024; k0 += 32) {
          acc = wmma_f16(fragA8(sComb, 1024, k0), fragB(W, 1024, k0, n0), acc);
        }
        _Float16* dst = (wid < 4) ? sT1 : sF1;
        const int nl = lane & 15, mb = (lane >> 4) * 8;
        const int n = n0 + nl;
#pragma unroll
        for (int v = 0; v < 8; ++v) {
          const int r = mb + v;
          if (r < 8) dst[r * 64 + n] = (_Float16)tanhf(acc[v] + Bv[n]);
        }
      }
      __syncthreads();

      // stage 2: [16,64] @ [64,512]; 32 fragments per branch, 8 per wave
      v8f facc[8];
      {
        const _Float16* W = (wid < 4) ? tau2_wc : bb2_wc;
        const _Float16* src = (wid < 4) ? sT1 : sF1;
#pragma unroll
        for (int j = 0; j < 8; ++j) {
          const int n0 = ((wid & 3) * 8 + j) * 16;
#pragma unroll
          for (int v = 0; v < 8; ++v) facc[j][v] = 0.f;
#pragma unroll
          for (int ks = 0; ks < 2; ++ks) {
            facc[j] = wmma_f16(fragA8(src, 64, ks * 32),
                               fragB(W, 64, ks * 32, n0), facc[j]);
          }
        }
      }
      if (wid < 4) {  // tau branch -> sTau
        const int nl = lane & 15, mb = (lane >> 4) * 8;
#pragma unroll
        for (int j = 0; j < 8; ++j) {
          const int n = ((wid & 3) * 8 + j) * 16 + nl;
#pragma unroll
          for (int v = 0; v < 8; ++v) {
            const int r = mb + v;
            if (r < 8) {
              const float z = facc[j][v] + tau2_b[n];
              sTau[r * 512 + n] = 0.1f + 9.9f / (1.f + expf(-z));
            }
          }
        }
      }
      __syncthreads();
      if (wid >= 4) {  // backbone branch: ODE update in place
        const int nl = lane & 15, mb = (lane >> 4) * 8;
#pragma unroll
        for (int j = 0; j < 8; ++j) {
          const int n = ((wid & 3) * 8 + j) * 16 + nl;
#pragma unroll
          for (int v = 0; v < 8; ++v) {
            const int r = mb + v;
            if (r < 8) {
              const float fv = facc[j][v] + bb2_b[n];
              const float sv = sState[r * 512 + n];
              const float gv = g[(r * Ss + t) * Hh + n];
              const float tv = sTau[r * 512 + n];
              sState[r * 512 + n] = sv + dt * (-sv + fv * gv) / tv;
            }
          }
        }
      }
      __syncthreads();
    }
#pragma unroll
    for (int it = 0; it < 4; ++it) {
      const int i = tid * 4 + it * 1024;
      const int bi = i >> 9, c = i & 511;
      float4 f;
      f.x = sState[i + 0]; f.y = sState[i + 1];
      f.z = sState[i + 2]; f.w = sState[i + 3];
      *(float4*)(ys + (bi * Ss + t) * Hh + c) = f;
    }
    __syncthreads();
  }
}

// ---------------------------------------------------------------------------
// Host orchestration
// ---------------------------------------------------------------------------
extern "C" void kernel_launch(void* const* d_in, const int* in_sizes, int n_in,
                              void* d_out, int out_size, void* d_ws,
                              size_t ws_size, hipStream_t stream) {
  (void)in_sizes; (void)n_in; (void)out_size; (void)ws_size;

  const int*   ids    = (const int*)  d_in[0];
  const float* tok    = (const float*)d_in[1];
  const float* pos    = (const float*)d_in[2];
  const float* bb1_w  = (const float*)d_in[3];
  const float* bb1_b  = (const float*)d_in[4];
  const float* bb2_w  = (const float*)d_in[5];
  const float* bb2_b  = (const float*)d_in[6];
  const float* tau1_w = (const float*)d_in[7];
  const float* tau1_b = (const float*)d_in[8];
  const float* tau2_w = (const float*)d_in[9];
  const float* tau2_b = (const float*)d_in[10];
  const float* sens_w = (const float*)d_in[11];
  const float* sens_b = (const float*)d_in[12];
  const float* q_w    = (const float*)d_in[13];
  const float* q_b    = (const float*)d_in[14];
  const float* k_w    = (const float*)d_in[15];
  const float* k_b    = (const float*)d_in[16];
  const float* v_w    = (const float*)d_in[17];
  const float* v_b    = (const float*)d_in[18];
  const float* o_w    = (const float*)d_in[19];
  const float* o_b    = (const float*)d_in[20];
  const float* gate_w = (const float*)d_in[21];
  const float* gate_b = (const float*)d_in[22];
  const float* e_w1   = (const float*)d_in[23];
  const float* e_b1   = (const float*)d_in[24];
  const float* e_w2   = (const float*)d_in[25];
  const float* e_b2   = (const float*)d_in[26];
  const float* n1_g   = (const float*)d_in[27];
  const float* n1_b   = (const float*)d_in[28];
  const float* n2_g   = (const float*)d_in[29];
  const float* n2_b   = (const float*)d_in[30];
  const float* n3_g   = (const float*)d_in[31];
  const float* n3_b   = (const float*)d_in[32];
  const float* fn_g   = (const float*)d_in[33];
  const float* fn_b   = (const float*)d_in[34];

  const size_t T = (size_t)Bb * Ss;  // 2048 tokens
  float* ws   = (float*)d_ws;
  float* xb   = ws;                  // [T,H] residual stream
  float* yb   = xb   + T * Hh;       // [T,H]
  float* gb   = yb   + T * Hh;       // [T,H] LTC sensory gate
  float* ltcb = gb   + T * Hh;       // [T,H]
  float* qb   = ltcb + T * Hh;
  float* kb   = qb   + T * Hh;
  float* vb   = kb   + T * Hh;
  float* ab   = vb   + T * Hh;       // attention out
  float* pb   = ab   + T * Hh;       // o-proj out
  float* accb = pb   + T * Hh;       // MoE accumulator
  float* hb   = accb + T * Hh;       // [T,F] expert hidden
  float* wgt  = hb   + T * Ff;       // [T,E] combine weights
  // f16 column-major scan weights (per layer, reconverted each layer)
  _Float16* w1t = (_Float16*)(wgt + T * Ee);   // tau1: [64][1024]
  _Float16* w1b = w1t + 1024 * 64;             // bb1:  [64][1024]
  _Float16* w2t = w1b + 1024 * 64;             // tau2: [512][64]
  _Float16* w2b = w2t + 64 * 512;              // bb2:  [512][64]

  const dim3 blk(256);
  auto gemm = [&](const float* A, const float* Bm, const float* bias, float* C,
                  int M, int N, int K, int act, const float* rs, int rss,
                  int accum, int btr) {
    dim3 grid((N + 127) / 128, (M + 63) / 64);
    wmma_gemm_kernel<<<grid, blk, 0, stream>>>(A, Bm, bias, C, M, N, K, act,
                                               rs, rss, accum, btr);
  };

  embed_kernel<<<(int)(T * Hh / 256), blk, 0, stream>>>(ids, tok, pos, xb);

  for (int l = 0; l < 2; ++l) {
    // LTC sensory gate: sigmoid(x @ sens_w + sens_b)
    gemm(xb, sens_w + (size_t)l * Hh * Hh, sens_b + l * Hh, gb,
         (int)T, Hh, Hh, 1, nullptr, 0, 0, 0);
    // convert scan weights to f16 column-major
    conv_w_kernel<<<1024 * 64 / 256, blk, 0, stream>>>(
        tau1_w + (size_t)l * 2 * Hh * BUu, w1t, 1024, 64);
    conv_w_kernel<<<1024 * 64 / 256, blk, 0, stream>>>(
        bb1_w + (size_t)l * 2 * Hh * BUu, w1b, 1024, 64);
    conv_w_kernel<<<64 * 512 / 256, blk, 0, stream>>>(
        tau2_w + (size_t)l * BUu * Hh, w2t, 64, 512);
    conv_w_kernel<<<64 * 512 / 256, blk, 0, stream>>>(
        bb2_w + (size_t)l * BUu * Hh, w2b, 64, 512);
    // sequential ODE scan (one persistent workgroup)
    ltc_scan_kernel<<<1, blk, 0, stream>>>(
        xb, gb,
        w1t, tau1_b + l * BUu, w2t, tau2_b + l * Hh,
        w1b, bb1_b + l * BUu, w2b, bb2_b + l * Hh,
        ltcb);
    ln_kernel<<<(int)T, blk, 0, stream>>>(xb, ltcb, n1_g + l * Hh, n1_b + l * Hh, yb);

    // attention
    gemm(yb, q_w + (size_t)l * Hh * Hh, q_b + l * Hh, qb, (int)T, Hh, Hh, 0, nullptr, 0, 0, 0);
    gemm(yb, k_w + (size_t)l * Hh * Hh, k_b + l * Hh, kb, (int)T, Hh, Hh, 0, nullptr, 0, 0, 0);
    gemm(yb, v_w + (size_t)l * Hh * Hh, v_b + l * Hh, vb, (int)T, Hh, Hh, 0, nullptr, 0, 0, 0);
    attn_kernel<<<dim3(Ss / 64, NHh, Bb), blk, 0, stream>>>(qb, kb, vb, ab);
    gemm(ab, o_w + (size_t)l * Hh * Hh, o_b + l * Hh, pb, (int)T, Hh, Hh, 0, nullptr, 0, 0, 0);
    ln_kernel<<<(int)T, blk, 0, stream>>>(yb, pb, n2_g + l * Hh, n2_b + l * Hh, xb);

    // MoE (all experts evaluated densely; top-2 combine weights)
    moe_gate_kernel<<<(int)(T / 256), blk, 0, stream>>>(
        xb, gate_w + (size_t)l * Hh * Ee, gate_b + l * Ee, wgt);
    zero_kernel<<<(int)(T * Hh / 256), blk, 0, stream>>>(accb, (int)(T * Hh));
    for (int e = 0; e < Ee; ++e) {
      const size_t we = (size_t)(l * Ee + e);
      gemm(xb, e_w1 + we * Hh * Ff, e_b1 + we * Ff, hb,
           (int)T, Ff, Hh, 2 /*gelu*/, nullptr, 0, 0, 0);
      gemm(hb, e_w2 + we * Ff * Hh, e_b2 + we * Hh, accb,
           (int)T, Hh, Ff, 0, wgt + e, Ee, 1 /*accum*/, 0);
    }
    ln_kernel<<<(int)T, blk, 0, stream>>>(xb, accb, n3_g + l * Hh, n3_b + l * Hh, xb);
  }

  ln_kernel<<<(int)T, blk, 0, stream>>>(xb, nullptr, fn_g, fn_b, yb);
  // tied lm_head: logits = y @ tok_emb^T  (B transposed: tok_emb is [V,H])
  gemm(yb, tok, nullptr, (float*)d_out, (int)T, Vv, Hh, 0, nullptr, 0, 0, 1);
}
